// MoE_79706003079244
// MI455X (gfx1250) — matrix operand order repaired and checked by reading, not compiled, not run on hardware
//
#include <hip/hip_runtime.h>

// Problem constants (match reference: B,S,D,E,K,FF = 4,2048,1024,8,2,4096)
#define NTOK   8192
#define DMODEL 1024
#define NEXP   8
#define FFDIM  4096
#define TM     32      // tokens per block (2 WMMA M-tiles)
#define CHUNK  2048    // FF columns of the silu intermediate held in LDS at a time

typedef __attribute__((ext_vector_type(16))) __bf16       v16bf;
typedef __attribute__((ext_vector_type(8)))  float        v8f;
typedef __attribute__((ext_vector_type(4)))  unsigned int u32x4;

union Frag { v16bf f; u32x4 u[2]; };

__device__ __forceinline__ unsigned short f2bf(float f) {
  unsigned int u = __float_as_uint(f);
  u += 0x7FFFu + ((u >> 16) & 1u);   // round-to-nearest-even
  return (unsigned short)(u >> 16);
}

// dst[e][c][r] (bf16) = src[e][r][c] (f32). For w1: R=DMODEL,C=FFDIM. For w2: R=FFDIM,C=DMODEL.
__global__ __launch_bounds__(256)
void transpose_cast_kernel(const float* __restrict__ src, unsigned short* __restrict__ dst,
                           int R, int C) {
  unsigned long long total = (unsigned long long)NEXP * R * C;
  unsigned long long step  = (unsigned long long)gridDim.x * 256ull;
  for (unsigned long long o = (unsigned long long)blockIdx.x * 256ull + threadIdx.x;
       o < total; o += step) {
    unsigned long long rc = (unsigned long long)R * C;
    unsigned long long e  = o / rc;
    unsigned long long rem = o - e * rc;
    unsigned long long c  = rem / R;
    unsigned long long r  = rem - c * R;
    dst[o] = f2bf(src[(e * R + r) * C + c]);   // consecutive o -> consecutive r: coalesced store
  }
}

// Gating: logits = x @ gate_w [N,8]; top-2; softmax over the 2; scatter into combine[N,8].
__global__ __launch_bounds__(256)
void gate_kernel(const float* __restrict__ x, const float* __restrict__ gw,
                 float* __restrict__ combine) {
  __shared__ float gws[DMODEL * NEXP];                 // 32 KB
  for (int i = threadIdx.x; i < DMODEL * NEXP; i += 256) gws[i] = gw[i];
  __syncthreads();

  int t = blockIdx.x * 256 + threadIdx.x;
  const float* xp = x + (unsigned long long)t * DMODEL;
  float logit[NEXP];
#pragma unroll
  for (int e = 0; e < NEXP; ++e) logit[e] = 0.f;
  for (int d = 0; d < DMODEL; ++d) {
    float xv = xp[d];                                  // one row per thread: lanes read DMODEL floats apart
#pragma unroll
    for (int e = 0; e < NEXP; ++e) logit[e] = fmaf(xv, gws[d * NEXP + e], logit[e]);
  }
  int i1 = 0; float v1 = logit[0];
#pragma unroll
  for (int e = 1; e < NEXP; ++e) if (logit[e] > v1) { v1 = logit[e]; i1 = e; }
  int i2 = -1; float v2 = -3.0e38f;
#pragma unroll
  for (int e = 0; e < NEXP; ++e) if (e != i1 && logit[e] > v2) { v2 = logit[e]; i2 = e; }
  float e2 = __expf(v2 - v1);
  float inv = 1.f / (1.f + e2);
  float wa = inv, wb = e2 * inv;                       // softmax over top-2 (v1 is max)
#pragma unroll
  for (int e = 0; e < NEXP; ++e)
    combine[(unsigned long long)t * NEXP + e] = (e == i1) ? wa : ((e == i2) ? wb : 0.f);
}

// Fused MoE over one 32-token tile. Dense over experts, weighted by combine (zero for
// unselected experts -> same math as the reference loop). Weights stream from the
// L2-resident bf16 transposed copies; each weight fragment is loaded ONCE per block and
// shared by the two M-tiles. The combine weight is folded into hs so GEMM2's WMMAs
// accumulate directly into the persistent register accumulator (C operand = oacc).
__global__ __launch_bounds__(256)
void moe_fused_kernel(const float* __restrict__ x,
                      const unsigned short* __restrict__ w1t,  // [E][FF][D] bf16
                      const float* __restrict__ b1,            // [E][FF]
                      const unsigned short* __restrict__ w2t,  // [E][D][FF] bf16
                      const float* __restrict__ b2,            // [E][D]
                      const float* __restrict__ combine,       // [N][E]
                      float* __restrict__ out) {
  extern __shared__ unsigned short smem[];
  unsigned short* xs = smem;                       // [TM][DMODEL] bf16  (64 KB)
  unsigned short* hs = smem + TM * DMODEL;         // [TM][CHUNK] bf16   (128 KB)

  const int tid  = threadIdx.x;
  const int lane = tid & 31;
  const int wave = tid >> 5;                       // 8 waves
  const int tok0 = blockIdx.x * TM;
  const int mrow  = lane & 15;                     // M (A-frag) / N (B-frag) / N (C) per ISA layouts
  const int khalf = (lane >> 4) << 3;              // 0/8: A-frag K sub-offset; also C row-half offset
  const int kq    = (lane >> 4) << 4;              // 0/16: unused (B-frag K sub-offset is khalf)

  // Stage x tile as bf16 into LDS.
  for (int i = tid; i < TM * DMODEL; i += 256) {
    int m = i >> 10, k = i & (DMODEL - 1);
    xs[i] = f2bf(x[(unsigned long long)(tok0 + m) * DMODEL + k]);
  }

  v8f oacc[16];                                     // 32 rows x 1024 cols / (8 waves * 32 lanes)
#pragma unroll
  for (int i = 0; i < 16; ++i) oacc[i] = {};

  for (int e = 0; e < NEXP; ++e) {
    const unsigned short* w1e = w1t + (unsigned long long)e * FFDIM * DMODEL;
    const unsigned short* w2e = w2t + (unsigned long long)e * DMODEL * FFDIM;
    const float* b1e = b1 + e * FFDIM;
    const float* b2e = b2 + e * DMODEL;

    // Per-thread combine weights for the rows this lane touches (small cached region).
    float cwr0[8], cwr1[8];
#pragma unroll
    for (int g = 0; g < 8; ++g) {
      cwr0[g] = combine[(unsigned long long)(tok0 + g + khalf) * NEXP + e];
      cwr1[g] = combine[(unsigned long long)(tok0 + 16 + g + khalf) * NEXP + e];
    }

    for (int ch = 0; ch < FFDIM / CHUNK; ++ch) {
      __syncthreads();                              // hs drained by previous GEMM2

      // ---- GEMM1: hs = cw[m] * silu(x @ w1[e] + b1[e]) for columns [ch*CHUNK, +CHUNK) ----
      for (int nt = wave; nt < CHUNK / 16; nt += 8) {
        const int ncol = ch * CHUNK + (nt << 4);
        v8f acc0 = {}, acc1 = {};
        const unsigned short* arow0 = xs + mrow * DMODEL;            // rows 0..15
        const unsigned short* arow1 = xs + (16 + mrow) * DMODEL;     // rows 16..31
        const unsigned short* bcol  = w1e + (unsigned long long)(ncol + mrow) * DMODEL + khalf;
#pragma unroll 2
        for (int ks = 0; ks < DMODEL; ks += 32) {
          Frag b, a0, a1;
          b.u[0]  = *(const u32x4*)(bcol + ks);
          b.u[1]  = *(const u32x4*)(bcol + ks + 16);
          a0.u[0] = *(const u32x4*)(arow0 + ks + khalf);
          a0.u[1] = *(const u32x4*)(arow0 + ks + 16 + khalf);
          a1.u[0] = *(const u32x4*)(arow1 + ks + khalf);
          a1.u[1] = *(const u32x4*)(arow1 + ks + 16 + khalf);
          acc0 = __builtin_amdgcn_wmma_f32_16x16x32_bf16(false, a0.f, false, b.f,
                                                         (short)0, acc0, false, false);
          acc1 = __builtin_amdgcn_wmma_f32_16x16x32_bf16(false, a1.f, false, b.f,
                                                         (short)0, acc1, false, false);
        }
        const float bv = b1e[ncol + mrow];
#pragma unroll
        for (int g = 0; g < 8; ++g) {
          float v0 = acc0[g] + bv;
          float v1 = acc1[g] + bv;
          // silu via fast reciprocal (single v_rcp_f32), pre-scaled by combine weight
          float s0 = cwr0[g] * v0 * __builtin_amdgcn_rcpf(1.f + __expf(-v0));
          float s1 = cwr1[g] * v1 * __builtin_amdgcn_rcpf(1.f + __expf(-v1));
          hs[(g + khalf) * CHUNK + (nt << 4) + mrow]      = f2bf(s0);
          hs[(16 + g + khalf) * CHUNK + (nt << 4) + mrow] = f2bf(s1);
        }
      }
      __syncthreads();

      // ---- GEMM2: oacc += (cw ⊙ h) @ w2[e][ch-slice]  (WMMA C operand = oacc) ----
#pragma unroll
      for (int p = 0; p < 8; ++p) {                 // fully unrolled: static oacc indices
        const int dcol = (wave + 8 * p) << 4;
        const int ti = 2 * p;
        const unsigned short* arow0 = hs + mrow * CHUNK;
        const unsigned short* arow1 = hs + (16 + mrow) * CHUNK;
        const unsigned short* bcol  = w2e + (unsigned long long)(dcol + mrow) * FFDIM
                                          + ch * CHUNK + khalf;
#pragma unroll 2
        for (int ks = 0; ks < CHUNK; ks += 32) {
          Frag b, a0, a1;
          b.u[0]  = *(const u32x4*)(bcol + ks);
          b.u[1]  = *(const u32x4*)(bcol + ks + 16);
          a0.u[0] = *(const u32x4*)(arow0 + ks + khalf);
          a0.u[1] = *(const u32x4*)(arow0 + ks + 16 + khalf);
          a1.u[0] = *(const u32x4*)(arow1 + ks + khalf);
          a1.u[1] = *(const u32x4*)(arow1 + ks + 16 + khalf);
          oacc[ti + 0] = __builtin_amdgcn_wmma_f32_16x16x32_bf16(false, a0.f, false, b.f,
                                                                 (short)0, oacc[ti + 0],
                                                                 false, false);
          oacc[ti + 1] = __builtin_amdgcn_wmma_f32_16x16x32_bf16(false, a1.f, false, b.f,
                                                                 (short)0, oacc[ti + 1],
                                                                 false, false);
        }
      }
    }

    // ---- bias term: oacc += cw[m] * b2[e][d]  (once per expert) ----
#pragma unroll
    for (int p = 0; p < 8; ++p) {
      const int dcol = (wave + 8 * p) << 4;
      const int ti = 2 * p;
      const float bv = b2e[dcol + mrow];
#pragma unroll
      for (int g = 0; g < 8; ++g) {
        oacc[ti + 0][g] = fmaf(cwr0[g], bv, oacc[ti + 0][g]);
        oacc[ti + 1][g] = fmaf(cwr1[g], bv, oacc[ti + 1][g]);
      }
    }
  }

  // Store the accumulated output tile (fp32).
#pragma unroll
  for (int p = 0; p < 8; ++p) {
    const int dcol = (wave + 8 * p) << 4;
    const int ti = 2 * p;
#pragma unroll
    for (int g = 0; g < 8; ++g) {
      out[(unsigned long long)(tok0 + g + khalf) * DMODEL + dcol + mrow]      = oacc[ti + 0][g];
      out[(unsigned long long)(tok0 + 16 + g + khalf) * DMODEL + dcol + mrow] = oacc[ti + 1][g];
    }
  }
}

extern "C" void kernel_launch(void* const* d_in, const int* in_sizes, int n_in,
                              void* d_out, int out_size, void* d_ws, size_t ws_size,
                              hipStream_t stream) {
  const float* x      = (const float*)d_in[0];   // [B,S,D]
  const float* gate_w = (const float*)d_in[1];   // [D,E]
  const float* w1     = (const float*)d_in[2];   // [E,D,FF]
  const float* b1     = (const float*)d_in[3];   // [E,FF]
  const float* w2     = (const float*)d_in[4];   // [E,FF,D]
  const float* b2     = (const float*)d_in[5];   // [E,D]
  float* out = (float*)d_out;

  // Workspace layout: w1t (64MB bf16) | w2t (64MB bf16) | combine (256KB f32)
  unsigned short* w1t = (unsigned short*)d_ws;
  unsigned short* w2t = w1t + (unsigned long long)NEXP * DMODEL * FFDIM;
  float* combine = (float*)(w2t + (unsigned long long)NEXP * DMODEL * FFDIM);

  // 1+2) one-time cast+transpose so the whole weight set (128MB bf16) lives in the 192MB L2
  transpose_cast_kernel<<<4096, 256, 0, stream>>>(w1, w1t, DMODEL, FFDIM);
  transpose_cast_kernel<<<4096, 256, 0, stream>>>(w2, w2t, FFDIM, DMODEL);

  // 3) gating -> combine[N,E]
  gate_kernel<<<NTOK / 256, 256, 0, stream>>>(x, gate_w, combine);

  // 4) fused MoE over 32-token tiles
  size_t smem = (size_t)(TM * DMODEL + TM * CHUNK) * sizeof(unsigned short); // 192 KB of 320 KB/WGP
  hipFuncSetAttribute((const void*)moe_fused_kernel,
                      hipFuncAttributeMaxDynamicSharedMemorySize, (int)smem);
  moe_fused_kernel<<<NTOK / TM, 256, smem, stream>>>(x, w1t, b1, w2t, b2, combine, out);
}
